// Mutil_Head_Attention_32315333935615
// MI455X (gfx1250) — compile-verified, matrix operand order checked by reading, not run on hardware
//
#include <hip/hip_runtime.h>
#include <hip/hip_bf16.h>

// ---- problem constants (match reference) ----
constexpr int Bb = 2, Ss = 2048, DMm = 1024, Hh = 16, Dd = 64;
constexpr float SCALE = 0.125f; // 1/sqrt(64)

typedef __attribute__((ext_vector_type(16))) __bf16 bf16x16;
typedef __attribute__((ext_vector_type(8)))  __bf16 bf16x8;
typedef __attribute__((ext_vector_type(8)))  float  f32x8;
typedef __attribute__((ext_vector_type(4)))  unsigned int u32x4;
typedef __attribute__((ext_vector_type(8)))  int i32x8;
typedef __attribute__((ext_vector_type(4)))  int i32x4;

// Load one 16x32 bf16 WMMA operand fragment from a row-major [rows, ld] matrix
// (global or LDS). Matches CDNA5 16-bit A-matrix layout:
//   lanes 0-15: M=lane, K = k0+{0..7} and k0+{16..23}
//   lanes 16-31: M=lane-16, K = k0+{8..15} and k0+{24..31}
__device__ __forceinline__ bf16x16 load_frag16(const __bf16* __restrict__ base,
                                               int ld, int row0, int k0) {
  const int lane = (int)(threadIdx.x & 31u);
  const int r  = row0 + (lane & 15);
  const int kb = k0 + ((lane >> 4) << 3);
  const __bf16* p = base + (size_t)r * ld + kb;      // 16B-aligned (kb % 8 == 0)
  const bf16x8 lo = *(const bf16x8*)(p);
  const bf16x8 hi = *(const bf16x8*)(p + 16);
  bf16x16 f;
#pragma unroll
  for (int i = 0; i < 8; ++i) { f[i] = lo[i]; f[i + 8] = hi[i]; }
  return f;
}

// Issue one TDM 2-D tile DMA: global (row-major, element=2B) -> LDS (contiguous).
// Tile = rows_tile x 32 elements at gptr; tensor dims for OOB; D# per ISA 8.3/8.4.
// 6-arg builtin form (clang-23 / therock-10.0): groups 2,3 + extra group zeroed.
__device__ __forceinline__ void tdm_load_tile(const __bf16* gptr, unsigned lds_off,
                                              int rows_tile, int tensor_d0,
                                              int tensor_d1, int stride_elems) {
  const unsigned long long ga = (unsigned long long)(uintptr_t)gptr;
  u32x4 g0;
  g0[0] = 1u;                                   // count=1, user mode, no gather
  g0[1] = lds_off;                              // lds_addr (bytes)
  g0[2] = (unsigned)(ga & 0xFFFFFFFFu);         // global_addr[31:0]
  g0[3] = (unsigned)((ga >> 32) & 0x01FFFFFFu)  // global_addr[56:32]
        | (2u << 30);                           // type=2 (image)
  i32x8 g1;
  g1[0] = 0x00010000;                           // data_size=1 (2 bytes), mask=0
  g1[1] = (int)(((unsigned)tensor_d0 & 0xFFFFu) << 16);          // dim0 lo16
  g1[2] = (int)((((unsigned)tensor_d0 >> 16) & 0xFFFFu)
        | (((unsigned)tensor_d1 & 0xFFFFu) << 16));              // dim0 hi|dim1 lo
  g1[3] = (int)((((unsigned)tensor_d1 >> 16) & 0xFFFFu)
        | (32u << 16));                                          // tile_dim0=32
  g1[4] = rows_tile;                            // tile_dim1 (tile_dim2=0)
  g1[5] = stride_elems;                         // tensor_dim0_stride[31:0]
  g1[6] = 0;
  g1[7] = 0;
  const i32x4 z4 = {0, 0, 0, 0};
  const i32x8 z8 = {0, 0, 0, 0, 0, 0, 0, 0};
  __builtin_amdgcn_tensor_load_to_lds(g0, g1, z4, z4, z8, 0);
}

// ---------------- elementwise fp32 -> bf16 convert ----------------
__global__ void cvt_f32_to_bf16(const float* __restrict__ x,
                                __bf16* __restrict__ y, size_t n) {
  const size_t i = ((size_t)blockIdx.x * blockDim.x + threadIdx.x) * 4;
  if (i + 3 < n) {
    const float4 f = *(const float4*)(x + i);
    y[i]     = (__bf16)f.x;
    y[i + 1] = (__bf16)f.y;
    y[i + 2] = (__bf16)f.z;
    y[i + 3] = (__bf16)f.w;
  }
}

// ------------- transpose + convert: WT[n][k] = (bf16)W[k][n] -------------
__global__ __launch_bounds__(256) void transpose_cvt(const float* __restrict__ W,
                                                     __bf16* __restrict__ WT,
                                                     int rows, int cols) {
  __shared__ float tile[32][33];
  const int r0 = blockIdx.y * 32, c0 = blockIdx.x * 32;
  const int tx = threadIdx.x & 31, ty = threadIdx.x >> 5; // 8 row-groups
#pragma unroll
  for (int i = ty; i < 32; i += 8)
    tile[i][tx] = W[(size_t)(r0 + i) * cols + (c0 + tx)];
  __syncthreads();
#pragma unroll
  for (int i = ty; i < 32; i += 8)
    WT[(size_t)(c0 + i) * rows + (r0 + tx)] = (__bf16)tile[tx][i];
}

// ------------- GEMM: C[M,N] = A[M,K](bf16) * BT[N,K](bf16)^T + bias -------------
// Block = 8 waves, tile 128(M) x 64(N). TDM double-buffers 128x32 A and 64x32 BT
// slabs into LDS (wave 0 issues DMA, S_WAIT_TENSORCNT + barrier hands off);
// all waves run WMMAs out of LDS.
__global__ __launch_bounds__(256) void gemm_bf16_wmma(
    const __bf16* __restrict__ A, const __bf16* __restrict__ BT,
    const float* __restrict__ bias, __bf16* __restrict__ Cb,
    float* __restrict__ Cf, int M, int N, int K) {
  __shared__ __align__(16) __bf16 sA[2][128 * 32];   // 2 x 8 KB
  __shared__ __align__(16) __bf16 sB[2][64 * 32];    // 2 x 4 KB

  const int lane = threadIdx.x & 31;
  const int wave = threadIdx.x >> 5;
  const int m0 = blockIdx.y * 128;
  const int n0 = blockIdx.x * 64;
  const int NK = K >> 5;

  // prologue: stage 0 into buffer 0
  if (wave == 0) {
    tdm_load_tile(A + (size_t)m0 * K, (unsigned)(uintptr_t)&sA[0][0], 128, K, M, K);
    tdm_load_tile(BT + (size_t)n0 * K, (unsigned)(uintptr_t)&sB[0][0], 64, K, N, K);
  }

  f32x8 acc[4] = {};
  for (int ks = 0; ks < NK; ++ks) {
    const int cur = ks & 1;
    if (wave == 0) {
      if (ks + 1 < NK) {
        const int kn = (ks + 1) << 5;
        tdm_load_tile(A + (size_t)m0 * K + kn,
                      (unsigned)(uintptr_t)&sA[cur ^ 1][0], 128, K, M, K);
        tdm_load_tile(BT + (size_t)n0 * K + kn,
                      (unsigned)(uintptr_t)&sB[cur ^ 1][0], 64, K, N, K);
        __builtin_amdgcn_s_wait_tensorcnt(2);   // stage ks complete (in-order)
      } else {
        __builtin_amdgcn_s_wait_tensorcnt(0);
      }
    }
    __syncthreads();                            // LDS slab visible to all waves

    const bf16x16 a = load_frag16(&sA[cur][0], 32, wave * 16, 0);
#pragma unroll
    for (int t = 0; t < 4; ++t) {
      const bf16x16 b = load_frag16(&sB[cur][0], 32, 16 * t, 0);
      acc[t] = __builtin_amdgcn_wmma_f32_16x16x32_bf16(
          false, a, false, b, (short)0, acc[t], false, false);
    }
    __syncthreads();                            // reads done before buffer reuse
  }

  const int roff = (lane >> 4) << 3;
#pragma unroll
  for (int t = 0; t < 4; ++t) {
    const int col = n0 + 16 * t + (lane & 15);
    const float bv = bias[col];
#pragma unroll
    for (int r = 0; r < 8; ++r) {
      const float v = acc[t][r] + bv;
      const size_t idx = (size_t)(m0 + wave * 16 + roff + r) * N + col;
      if (Cf) Cf[idx] = v;
      else    Cb[idx] = (__bf16)v;
    }
  }
}

// ------------- attention scores + softmax (flash-style two-pass) -------------
// grid = (S/16, H, B), block = 256 (8 waves). Each block: 16 q-rows x all 2048 keys.
// Wave w owns keys [w*256, w*256+256): 16 key tiles, 2 WMMAs each (K=64).
__global__ __launch_bounds__(256) void attn_scores_softmax(
    const __bf16* __restrict__ qp, const __bf16* __restrict__ kp,
    float* __restrict__ score) {
  const int lane = threadIdx.x & 31;
  const int wave = threadIdx.x >> 5;
  const int qt = blockIdx.x, h = blockIdx.y, b = blockIdx.z;

  const __bf16* qbase = qp + (size_t)b * Ss * DMm + h * Dd;
  const __bf16* kbase = kp + (size_t)b * Ss * DMm + h * Dd;
  float* srow = score + (((size_t)b * Hh + h) * Ss + (size_t)qt * 16) * Ss;

  // Q fragments reused across all 16 key tiles of this wave.
  const bf16x16 a0 = load_frag16(qbase, DMm, qt * 16, 0);
  const bf16x16 a1 = load_frag16(qbase, DMm, qt * 16, 32);

  const int roff = (lane >> 4) << 3;
  float m[8], s[8];
#pragma unroll
  for (int r = 0; r < 8; ++r) { m[r] = -1e30f; s[r] = 0.f; }

  const int key0 = wave * 256;
  // ---- pass 1: WMMA score tiles -> global (raw, scaled) + online stats ----
  for (int t = 0; t < 16; ++t) {
    const int kcol = key0 + t * 16;
    if (t + 1 < 16) {  // prefetch next key tile rows (global_prefetch_b8)
      const __bf16* np = kbase + (size_t)(kcol + 16 + (lane & 15)) * DMm
                       + ((lane >> 4) << 3);
      __builtin_prefetch(np, 0, 1);
    }
    const bf16x16 b0 = load_frag16(kbase, DMm, kcol, 0);
    const bf16x16 b1 = load_frag16(kbase, DMm, kcol, 32);
    f32x8 c = {};
    c = __builtin_amdgcn_wmma_f32_16x16x32_bf16(false, a0, false, b0, (short)0, c, false, false);
    c = __builtin_amdgcn_wmma_f32_16x16x32_bf16(false, a1, false, b1, (short)0, c, false, false);
    const int col = kcol + (lane & 15);
#pragma unroll
    for (int r = 0; r < 8; ++r) {
      const float x = c[r] * SCALE;
      srow[(size_t)(roff + r) * Ss + col] = x;
      const float mn = fmaxf(m[r], x);
      s[r] = s[r] * __expf(m[r] - mn) + __expf(x - mn);
      m[r] = mn;
    }
  }

  // reduce online stats across the 16 lanes that share each row
#pragma unroll
  for (int d = 1; d < 16; d <<= 1) {
#pragma unroll
    for (int r = 0; r < 8; ++r) {
      const float mo = __shfl_xor(m[r], d, 32);
      const float so = __shfl_xor(s[r], d, 32);
      const float mn = fmaxf(m[r], mo);
      s[r] = s[r] * __expf(m[r] - mn) + so * __expf(mo - mn);
      m[r] = mn;
    }
  }

  __shared__ float sm[8][16];
  __shared__ float ssum[8][16];
  if ((lane & 15) == 0) {
#pragma unroll
    for (int r = 0; r < 8; ++r) {
      sm[wave][roff + r] = m[r];
      ssum[wave][roff + r] = s[r];
    }
  }
  __syncthreads();

  __shared__ float fm[16], fis[16];
  if (threadIdx.x < 16) {
    float mm = -1e30f, sv = 0.f;
    for (int w = 0; w < 8; ++w) {
      const float mo = sm[w][threadIdx.x], so = ssum[w][threadIdx.x];
      const float mn = fmaxf(mm, mo);
      sv = sv * __expf(mm - mn) + so * __expf(mo - mn);
      mm = mn;
    }
    fm[threadIdx.x] = mm;
    fis[threadIdx.x] = 1.f / sv;
  }
  __syncthreads();

  // ---- pass 2: each wave renormalizes its own 256-col span (L2-resident) ----
  for (int r = 0; r < 16; ++r) {
    const float mr = fm[r], is = fis[r];
    float* p = srow + (size_t)r * Ss + key0;
#pragma unroll
    for (int j = 0; j < 8; ++j) {
      const float x = p[lane + 32 * j];
      p[lane + 32 * j] = __expf(x - mr) * is;
    }
  }
}

extern "C" void kernel_launch(void* const* d_in, const int* in_sizes, int n_in,
                              void* d_out, int out_size, void* d_ws, size_t ws_size,
                              hipStream_t stream) {
  (void)in_sizes; (void)n_in; (void)out_size; (void)ws_size;
  const float* q  = (const float*)d_in[0];
  const float* k  = (const float*)d_in[1];
  const float* v  = (const float*)d_in[2];
  const float* Wq = (const float*)d_in[3];
  const float* bq = (const float*)d_in[4];
  const float* Wk = (const float*)d_in[5];
  const float* bk = (const float*)d_in[6];
  const float* Wv = (const float*)d_in[7];
  const float* bv = (const float*)d_in[8];
  const float* Wo = (const float*)d_in[9];
  const float* bo = (const float*)d_in[10];

  constexpr size_t MR = (size_t)Bb * Ss;     // 4096 rows
  constexpr size_t XE = MR * DMm;            // activation elems
  constexpr size_t WE = (size_t)DMm * DMm;   // weight elems

  // workspace layout (bf16): qb kb vb | WqT WkT WvT WoT | qp kp vp  (~56 MB)
  __bf16* ws  = (__bf16*)d_ws;
  __bf16* qb  = ws;
  __bf16* kb  = qb + XE;
  __bf16* vb  = kb + XE;
  __bf16* WqT = vb + XE;
  __bf16* WkT = WqT + WE;
  __bf16* WvT = WkT + WE;
  __bf16* WoT = WvT + WE;
  __bf16* qpj = WoT + WE;
  __bf16* kpj = qpj + XE;
  __bf16* vpj = kpj + XE;

  // 1) convert activations to bf16
  {
    const dim3 blk(256), grd((unsigned)((XE / 4 + 255) / 256));
    cvt_f32_to_bf16<<<grd, blk, 0, stream>>>(q, qb, XE);
    cvt_f32_to_bf16<<<grd, blk, 0, stream>>>(k, kb, XE);
    cvt_f32_to_bf16<<<grd, blk, 0, stream>>>(v, vb, XE);
  }
  // 2) transpose+convert weights ([K,N] fp32 -> [N,K] bf16)
  {
    const dim3 blk(256), grd(DMm / 32, DMm / 32);
    transpose_cvt<<<grd, blk, 0, stream>>>(Wq, WqT, DMm, DMm);
    transpose_cvt<<<grd, blk, 0, stream>>>(Wk, WkT, DMm, DMm);
    transpose_cvt<<<grd, blk, 0, stream>>>(Wv, WvT, DMm, DMm);
    transpose_cvt<<<grd, blk, 0, stream>>>(Wo, WoT, DMm, DMm);
  }
  // 3) projections (bf16 out), TDM-pipelined WMMA GEMM
  {
    const dim3 blk(256), grd(DMm / 64, (unsigned)(MR / 128));
    gemm_bf16_wmma<<<grd, blk, 0, stream>>>(qb, WqT, bq, qpj, nullptr, (int)MR, DMm, DMm);
    gemm_bf16_wmma<<<grd, blk, 0, stream>>>(kb, WkT, bk, kpj, nullptr, (int)MR, DMm, DMm);
    gemm_bf16_wmma<<<grd, blk, 0, stream>>>(vb, WvT, bv, vpj, nullptr, (int)MR, DMm, DMm);
  }
  // 4) attention scores + softmax -> d_out[0 .. B*H*S*S)
  {
    const dim3 blk(256), grd(Ss / 16, Hh, Bb);
    attn_scores_softmax<<<grd, blk, 0, stream>>>(qpj, kpj, (float*)d_out);
  }
  // 5) output projection (reference bug: merged == vp) -> fp32 out
  {
    float* outp = (float*)d_out + (size_t)Bb * Hh * Ss * Ss;
    const dim3 blk(256), grd(DMm / 64, (unsigned)(MR / 128));
    gemm_bf16_wmma<<<grd, blk, 0, stream>>>(vpj, WoT, bo, nullptr, outp, (int)MR, DMm, DMm);
  }
}
